// CliffordVideoAttention_63050119905980
// MI455X (gfx1250) — compile-verified, matrix operand order checked by reading, not run on hardware
//
#include <hip/hip_runtime.h>
#include <hip/hip_bf16.h>

typedef __bf16 bf16_t;
typedef __attribute__((ext_vector_type(16))) __bf16 v16bf;
typedef __attribute__((ext_vector_type(8)))  float  v8f;

#define L_      8192
#define INNER   2048
#define H_      16
#define D_      128
#define T_      8
#define S_      1024
#define NS_     16
#define NGATE   256

__constant__ int c_sp[12] = {0, 1, -1, 4, -4, 16, -16, 64, -64, 256, -256, 1024};
__constant__ int c_tp[4]  = {1, -1, 2, -2};

// ---------------------------------------------------------------- casts
__global__ void cast_f32_bf16(const float* __restrict__ in, bf16_t* __restrict__ out, long n) {
    long i = (long)blockIdx.x * blockDim.x + threadIdx.x;
    long stride = (long)gridDim.x * blockDim.x;
    for (; i < n; i += stride) out[i] = (bf16_t)in[i];
}

// in: [Kdim][Ndim] f32 row-major -> out: [Ndim][Kdim] bf16 row-major (B^T), LDS-tiled.
// grid: (Kdim/32, Ndim/32), block 256 = 32x8
__global__ __launch_bounds__(256) void transpose_cast(const float* __restrict__ in,
                                                      bf16_t* __restrict__ out,
                                                      int Kdim, int Ndim) {
    __shared__ float tile[32][33];
    const int kb = blockIdx.x * 32, nb = blockIdx.y * 32;
    const int tx = threadIdx.x & 31, ty = threadIdx.x >> 5;
#pragma unroll
    for (int i = 0; i < 4; ++i)
        tile[ty + 8 * i][tx] = in[(size_t)(kb + ty + 8 * i) * Ndim + nb + tx];
    __syncthreads();
#pragma unroll
    for (int i = 0; i < 4; ++i)
        out[(size_t)(nb + ty + 8 * i) * Kdim + kb + tx] = (bf16_t)tile[tx][ty + 8 * i];
}

// ---------------------------------------------------------------- WMMA GEMM
// C[M,N] = A[M,K](bf16) * B[K,N] + bias[N], Bt = B^T [N][K] bf16.
// Block tile 128x128, K-tile 64, 256 threads = 8 waves in 4x2 layout:
// wave (rg = wv>>1, cg = wv&1) computes rows [32*rg,32*rg+32) x cols [64*cg,64*cg+64)
// = 2x4 C tiles of 16x16. Double-buffered LDS; global->LDS copy with short-lived temps.
__device__ __forceinline__ void load_tile(bf16_t (&As)[128][64], bf16_t (&Bs)[128][64],
                                          const bf16_t* __restrict__ A,
                                          const bf16_t* __restrict__ Bt,
                                          int mBase, int nBase, int k0, int K, int tid) {
    const int row = tid >> 1;
    const int seg = (tid & 1) * 32;                 // 32 bf16 = 64B per thread per matrix
    const uint4* ga = (const uint4*)(A  + (size_t)(mBase + row) * K + k0 + seg);
    const uint4* gb = (const uint4*)(Bt + (size_t)(nBase + row) * K + k0 + seg);
    uint4 a0 = ga[0], a1 = ga[1], a2 = ga[2], a3 = ga[3];
    uint4 b0 = gb[0], b1 = gb[1], b2 = gb[2], b3 = gb[3];
    uint4* da = (uint4*)&As[row][seg];
    uint4* db = (uint4*)&Bs[row][seg];
    da[0] = a0; da[1] = a1; da[2] = a2; da[3] = a3;
    db[0] = b0; db[1] = b1; db[2] = b2; db[3] = b3;
}

__global__ __launch_bounds__(256) void gemm_bf16(const bf16_t* __restrict__ A,
                                                 const bf16_t* __restrict__ Bt,
                                                 const float* __restrict__ bias,
                                                 float* __restrict__ C,
                                                 int M, int N, int K) {
    __shared__ __align__(32) bf16_t As[2][128][64];     // 32KB
    __shared__ __align__(32) bf16_t Bs[2][128][64];     // 32KB
    const int tid  = threadIdx.x;
    const int wv   = tid >> 5;
    const int lane = tid & 31;
    const int mBase = blockIdx.y * 128;
    const int nBase = blockIdx.x * 128;
    const int rg = wv >> 1;                         // 0..3  (row group, 32 rows)
    const int cg = wv & 1;                          // 0..1  (col group, 64 cols)
    const int l15 = lane & 15;
    const int kHi = (lane >> 4) * 16;               // fragment K-half select

    v8f acc[8] = {};                                // [i*4+j]: i rows, j cols
    const int nk = K >> 6;                          // 64-wide K tiles

    load_tile(As[0], Bs[0], A, Bt, mBase, nBase, 0, K, tid);
    __syncthreads();

    for (int kt = 0; kt < nk; ++kt) {
        const int cur = kt & 1;
        if (kt + 1 < nk)
            load_tile(As[cur ^ 1], Bs[cur ^ 1], A, Bt, mBase, nBase, (kt + 1) << 6, K, tid);

#pragma unroll
        for (int c = 0; c < 2; ++c) {
            const int kOff = c * 32 + kHi;
            v16bf a0 = *(const v16bf*)&As[cur][rg * 32 +      l15][kOff];
            v16bf a1 = *(const v16bf*)&As[cur][rg * 32 + 16 + l15][kOff];
            v16bf b0 = *(const v16bf*)&Bs[cur][cg * 64 +      l15][kOff];
            v16bf b1 = *(const v16bf*)&Bs[cur][cg * 64 + 16 + l15][kOff];
            v16bf b2 = *(const v16bf*)&Bs[cur][cg * 64 + 32 + l15][kOff];
            v16bf b3 = *(const v16bf*)&Bs[cur][cg * 64 + 48 + l15][kOff];
            acc[0] = __builtin_amdgcn_wmma_f32_16x16x32_bf16(false, a0, false, b0, (short)0, acc[0], false, false);
            acc[1] = __builtin_amdgcn_wmma_f32_16x16x32_bf16(false, a0, false, b1, (short)0, acc[1], false, false);
            acc[2] = __builtin_amdgcn_wmma_f32_16x16x32_bf16(false, a0, false, b2, (short)0, acc[2], false, false);
            acc[3] = __builtin_amdgcn_wmma_f32_16x16x32_bf16(false, a0, false, b3, (short)0, acc[3], false, false);
            acc[4] = __builtin_amdgcn_wmma_f32_16x16x32_bf16(false, a1, false, b0, (short)0, acc[4], false, false);
            acc[5] = __builtin_amdgcn_wmma_f32_16x16x32_bf16(false, a1, false, b1, (short)0, acc[5], false, false);
            acc[6] = __builtin_amdgcn_wmma_f32_16x16x32_bf16(false, a1, false, b2, (short)0, acc[6], false, false);
            acc[7] = __builtin_amdgcn_wmma_f32_16x16x32_bf16(false, a1, false, b3, (short)0, acc[7], false, false);
        }
        __syncthreads();
    }

    // C layout: VGPR r -> M = r + 8*(lane>>4), N = lane&15
    const int rh = (lane >> 4) * 8;
#pragma unroll
    for (int j = 0; j < 4; ++j) {
        const int gn = nBase + cg * 64 + j * 16 + l15;
        const float bb = bias[gn];
#pragma unroll
        for (int i = 0; i < 2; ++i) {
            const int gm0 = mBase + rg * 32 + i * 16 + rh;
#pragma unroll
            for (int r = 0; r < 8; ++r)
                C[(size_t)(gm0 + r) * N + gn] = acc[i * 4 + j][r] + bb;
        }
    }
}

// ---------------------------------------------------------------- RMSNorm (Q): in-place f32 + bf16 copy
__global__ __launch_bounds__(256) void rmsnorm_q(float* __restrict__ q,
                                                 const float* __restrict__ w,
                                                 bf16_t* __restrict__ qb) {
    const int row = blockIdx.x, tid = threadIdx.x;
    const size_t base = (size_t)row * INNER;
    float vals[8];
    float ss = 0.f;
#pragma unroll
    for (int i = 0; i < 8; ++i) {
        vals[i] = q[base + tid + i * 256];
        ss += vals[i] * vals[i];
    }
    __shared__ float red[256];
    red[tid] = ss;
    __syncthreads();
    for (int s = 128; s > 0; s >>= 1) {
        if (tid < s) red[tid] += red[tid + s];
        __syncthreads();
    }
    const float rms = rsqrtf(red[0] * (1.0f / INNER) + 1e-6f);
#pragma unroll
    for (int i = 0; i < 8; ++i) {
        const int idx = tid + i * 256;
        const float o = vals[i] * rms * w[idx];
        q[base + idx]  = o;
        qb[base + idx] = (bf16_t)o;
    }
}

// ---------------------------------------------------------------- RMSNorm + channel-mix (K -> k_tilde, in place)
__global__ __launch_bounds__(256) void rmsnorm_kmix(float* __restrict__ k,
                                                    const float* __restrict__ w,
                                                    const float* __restrict__ mix) {
    const int row = blockIdx.x, tid = threadIdx.x;
    const size_t base = (size_t)row * INNER;
    __shared__ float rowsh[INNER];
    __shared__ float red[256];
    float vals[8];
    float ss = 0.f;
#pragma unroll
    for (int i = 0; i < 8; ++i) {
        vals[i] = k[base + tid + i * 256];
        ss += vals[i] * vals[i];
    }
    red[tid] = ss;
    __syncthreads();
    for (int s = 128; s > 0; s >>= 1) {
        if (tid < s) red[tid] += red[tid + s];
        __syncthreads();
    }
    const float rms = rsqrtf(red[0] * (1.0f / INNER) + 1e-6f);
#pragma unroll
    for (int i = 0; i < 8; ++i) {
        const int idx = tid + i * 256;
        rowsh[idx] = vals[i] * rms * w[idx];
    }
    __syncthreads();
    const float m0 = mix[0], m1 = mix[1], m2 = mix[2], m3 = mix[3], m4 = mix[4];
    const float SC = 0.08838834764831845f;   // 128^-0.5
#pragma unroll
    for (int i = 0; i < 8; ++i) {
        const int idx = tid + i * 256;
        const int b = idx & ~127, d = idx & 127;
        float kt = m0 * rowsh[idx]
                 + m1 * rowsh[b + ((d + 1) & 127)]
                 + m2 * rowsh[b + ((d + 2) & 127)]
                 + m3 * rowsh[b + ((d + 4) & 127)]
                 + m4 * rowsh[b + ((d + 8) & 127)];
        k[base + idx] = kt * SC;
    }
}

// ---------------------------------------------------------------- attention: 1 wave per (token, head)
__global__ __launch_bounds__(256) void attn_kernel(const float* __restrict__ q,
                                                   const float* __restrict__ kt,
                                                   const float* __restrict__ v,
                                                   const float* __restrict__ gl,
                                                   bf16_t* __restrict__ out) {
    __shared__ __align__(16) float qs[8][D_];
    __shared__ float wsh[8][NS_];
    __shared__ int   ssh[8][NS_];
    const int tid  = threadIdx.x;
    const int wv   = tid >> 5;
    const int lane = tid & 31;
    const int g    = blockIdx.x * 8 + wv;
    const int tok  = g >> 4;
    const int h    = g & 15;
    const size_t hoff  = (size_t)h * D_;
    const size_t qbase = (size_t)tok * INNER + hoff;

#pragma unroll
    for (int i = 0; i < 4; ++i) qs[wv][lane + 32 * i] = q[qbase + lane + 32 * i];
    __syncthreads();

    const int t = tok >> 10, s = tok & (S_ - 1);
    const int n = lane & 15;
    int src;
    if (n < 12) src = (t << 10) | ((s - c_sp[n]) & (S_ - 1));
    else        src = (((t - c_tp[n - 12]) & (T_ - 1)) << 10) | s;

    const int dh = (lane >> 4) * 64;
    const float4* qp = (const float4*)&qs[wv][dh];
    const float4* kp = (const float4*)(kt + (size_t)src * INNER + hoff + dh);
    float acc = 0.f;
#pragma unroll
    for (int i = 0; i < 16; ++i) {
        const float4 a = qp[i], b = kp[i];
        acc += a.x * b.x + a.y * b.y + a.z * b.z + a.w * b.w;
    }
    acc += __shfl_xor(acc, 16);
    float logit = acc + gl[(size_t)tok * (H_ * NS_) + h * NS_ + n];

    float mx = logit;
#pragma unroll
    for (int o = 1; o < 16; o <<= 1) mx = fmaxf(mx, __shfl_xor(mx, o));
    const float e = __expf(logit - mx);
    float sm = e;
#pragma unroll
    for (int o = 1; o < 16; o <<= 1) sm += __shfl_xor(sm, o);
    const float wgt = e / sm;

    if (lane < 16) { wsh[wv][lane] = wgt; ssh[wv][lane] = src; }
    __syncthreads();

    const int d0 = lane * 4;
    float4 o4 = {0.f, 0.f, 0.f, 0.f};
#pragma unroll
    for (int i = 0; i < NS_; ++i) {
        const float wn = wsh[wv][i];
        const float4 vv = *(const float4*)(v + (size_t)ssh[wv][i] * INNER + hoff + d0);
        o4.x += wn * vv.x; o4.y += wn * vv.y; o4.z += wn * vv.z; o4.w += wn * vv.w;
    }
    bf16_t* op = out + (size_t)tok * INNER + hoff + d0;
    op[0] = (bf16_t)o4.x; op[1] = (bf16_t)o4.y; op[2] = (bf16_t)o4.z; op[3] = (bf16_t)o4.w;
}

// ---------------------------------------------------------------- launch
extern "C" void kernel_launch(void* const* d_in, const int* in_sizes, int n_in,
                              void* d_out, int out_size, void* d_ws, size_t ws_size,
                              hipStream_t stream) {
    (void)in_sizes; (void)n_in; (void)out_size; (void)ws_size;
    const float* x      = (const float*)d_in[0];
    const float* Wq     = (const float*)d_in[1];
    const float* bq     = (const float*)d_in[2];
    const float* Wk     = (const float*)d_in[3];
    const float* bk     = (const float*)d_in[4];
    const float* Wv     = (const float*)d_in[5];
    const float* bv     = (const float*)d_in[6];
    const float* qn_w   = (const float*)d_in[7];
    const float* kn_w   = (const float*)d_in[8];
    const float* mix_w  = (const float*)d_in[9];
    const float* gate_W = (const float*)d_in[10];
    const float* gate_b = (const float*)d_in[11];
    const float* Wout   = (const float*)d_in[12];
    const float* bout   = (const float*)d_in[13];

    char* wp = (char*)d_ws;
    size_t off = 0;
    auto take = [&](size_t bytes) -> char* {
        char* p = wp + off;
        off = (off + bytes + 255) & ~(size_t)255;
        return p;
    };
    bf16_t* xb  = (bf16_t*)take((size_t)L_ * INNER * 2);
    bf16_t* wqT = (bf16_t*)take((size_t)INNER * INNER * 2);
    bf16_t* wkT = (bf16_t*)take((size_t)INNER * INNER * 2);
    bf16_t* wvT = (bf16_t*)take((size_t)INNER * INNER * 2);
    bf16_t* woT = (bf16_t*)take((size_t)INNER * INNER * 2);
    bf16_t* gwT = (bf16_t*)take((size_t)NGATE * INNER * 2);
    float*  qf  = (float*) take((size_t)L_ * INNER * 4);
    float*  kf  = (float*) take((size_t)L_ * INNER * 4);
    float*  vf  = (float*) take((size_t)L_ * INNER * 4);
    bf16_t* qb  = (bf16_t*)take((size_t)L_ * INNER * 2);
    float*  glb = (float*) take((size_t)L_ * NGATE * 4);
    bf16_t* ao  = (bf16_t*)take((size_t)L_ * INNER * 2);

    // casts / transposes
    cast_f32_bf16<<<2048, 256, 0, stream>>>(x, xb, (long)L_ * INNER);
    {
        dim3 gT(INNER / 32, INNER / 32);
        transpose_cast<<<gT, 256, 0, stream>>>(Wq,   wqT, INNER, INNER);
        transpose_cast<<<gT, 256, 0, stream>>>(Wk,   wkT, INNER, INNER);
        transpose_cast<<<gT, 256, 0, stream>>>(Wv,   wvT, INNER, INNER);
        transpose_cast<<<gT, 256, 0, stream>>>(Wout, woT, INNER, INNER);
        dim3 gG(INNER / 32, NGATE / 32);
        transpose_cast<<<gG, 256, 0, stream>>>(gate_W, gwT, INNER, NGATE);
    }

    // projections
    dim3 blk(256);
    dim3 gBig(INNER / 128, L_ / 128);      // (16, 64)
    gemm_bf16<<<gBig, blk, 0, stream>>>(xb, wqT, bq, qf, L_, INNER, INNER);
    gemm_bf16<<<gBig, blk, 0, stream>>>(xb, wkT, bk, kf, L_, INNER, INNER);
    gemm_bf16<<<gBig, blk, 0, stream>>>(xb, wvT, bv, vf, L_, INNER, INNER);

    // norms + premix
    rmsnorm_q<<<L_, 256, 0, stream>>>(qf, qn_w, qb);
    rmsnorm_kmix<<<L_, 256, 0, stream>>>(kf, kn_w, mix_w);

    // gate logits
    dim3 gGate(NGATE / 128, L_ / 128);     // (2, 64)
    gemm_bf16<<<gGate, blk, 0, stream>>>(qb, gwT, gate_b, glb, L_, NGATE, INNER);

    // shifted attention
    attn_kernel<<<(L_ * H_) / 8, 256, 0, stream>>>(qf, kf, vf, glb, ao);

    // output projection
    gemm_bf16<<<gBig, blk, 0, stream>>>(ao, woT, bout, (float*)d_out, L_, INNER, INNER);
}
